// Hybrid_e2e_stn_17308718203162
// MI455X (gfx1250) — matrix operand order repaired and checked by reading, not compiled, not run on hardware
//
#include <hip/hip_runtime.h>
#include <hip/hip_bf16.h>

// ---------------------------------------------------------------------------
// Types for CDNA5 WMMA
// ---------------------------------------------------------------------------
typedef __attribute__((ext_vector_type(16))) __bf16 bf16x16;
typedef __attribute__((ext_vector_type(8)))  float  f32x8;

union FragAB { unsigned int u[8]; bf16x16 v; };
union FragC  { float f[8]; f32x8 v; };

__device__ __forceinline__ unsigned short f2bf(float f) {
  unsigned int u = __float_as_uint(f);
  u += 0x7FFFu + ((u >> 16) & 1u);   // round-to-nearest-even
  return (unsigned short)(u >> 16);
}
__device__ __forceinline__ int imin(int a, int b) { return a < b ? a : b; }
__device__ __forceinline__ int imax(int a, int b) { return a > b ? a : b; }

__device__ __forceinline__ float gelu_tanh(float x) {
  const float c = 0.7978845608028654f;
  float t = tanhf(c * (x + 0.044715f * x * x * x));
  return 0.5f * x * (1.0f + t);
}

// CDNA5 async global->LDS copy (ASYNCcnt-tracked), see CDNA5 ISA
__device__ __forceinline__ void async_load_b128(unsigned lds_off, const void* gptr) {
  asm volatile("global_load_async_to_lds_b128 %0, %1, off"
               :: "v"(lds_off), "v"(gptr) : "memory");
}
__device__ __forceinline__ void wait_async0() {
  asm volatile("s_wait_asynccnt 0x0" ::: "memory");
}

// ---------------------------------------------------------------------------
// LDS-staged, double-buffered, batched GEMM.
//   C = act(alpha * A(bf16, MxK, row-major) @ BT(bf16, NxK)^T + bias + add + pos)
// Block: 8 waves arranged WMW x WNW; each wave computes a 32x32 output tile
// (2x2 of 16x16 WMMA tiles).  Block tile: (WMW*32) x (WNW*32).
// K must be a multiple of 32; M, N multiples of the block tile.
// Batch z: operand offset = (z/zdiv)*s1 + (z%zdiv)*s2.
// Per-thread staging descriptors are precomputed so the K-loop issues only
// the async copies, the ds_load_b128 fragment loads, and the WMMAs.
// ---------------------------------------------------------------------------
template <int WMW, int WNW>
__global__ void __launch_bounds__(256) gemm_bf16_lds_kernel(
    const unsigned short* __restrict__ A, long long sA1, long long sA2, int lda,
    const unsigned short* __restrict__ Bt, long long sB1, long long sB2, int ldb,
    float* C, long long sC1, long long sC2, int ldc,
    const float* __restrict__ bias, const float* addsrc,
    const float* __restrict__ pos,
    int M, int N, int K, int zdiv, float alpha, int relu)
{
  constexpr int BM  = WMW * 32;
  constexpr int BN  = WNW * 32;
  constexpr int APT = (BM * 4) / 256;   // A b128 chunks per thread (1 or 2)
  constexpr int BPT = (BN * 4) / 256;   // B b128 chunks per thread (2 or 1)
  __shared__ unsigned short As[2][BM][32];
  __shared__ unsigned short Bs[2][BN][32];
  constexpr unsigned AbufStride = BM * 32 * 2;  // bytes per LDS A buffer
  constexpr unsigned BbufStride = BN * 32 * 2;  // bytes per LDS B buffer

  const int nbn = N / BN;
  const int bm  = blockIdx.x / nbn;
  const int bn  = blockIdx.x % nbn;
  const int m0  = bm * BM, n0 = bn * BN;

  const int z = blockIdx.z;
  const long long aoff = (long long)(z / zdiv) * sA1 + (long long)(z % zdiv) * sA2;
  const long long boff = (long long)(z / zdiv) * sB1 + (long long)(z % zdiv) * sB2;
  const long long coff = (long long)(z / zdiv) * sC1 + (long long)(z % zdiv) * sC2;
  const unsigned short* Ab = A + aoff;
  const unsigned short* Bb = Bt + boff;

  const int tid  = threadIdx.x;
  const int wave = tid >> 5, lane = tid & 31;
  const int wm = wave / WNW, wn = wave % WNW;
  const int laneHi = lane >> 4, laneLo = lane & 15;

  // Precompute per-thread staging descriptors (global ptr + LDS offset).
  const unsigned short* gA[APT]; unsigned lA[APT];
#pragma unroll
  for (int i = 0; i < APT; ++i) {
    int idx = tid + i * 256, r = idx >> 2, seg = idx & 3;
    gA[i] = Ab + (long long)(m0 + r) * lda + seg * 8;
    lA[i] = (unsigned)(uintptr_t)&As[0][r][seg * 8];
  }
  const unsigned short* gB[BPT]; unsigned lB[BPT];
#pragma unroll
  for (int i = 0; i < BPT; ++i) {
    int idx = tid + i * 256, r = idx >> 2, seg = idx & 3;
    gB[i] = Bb + (long long)(n0 + r) * ldb + seg * 8;
    lB[i] = (unsigned)(uintptr_t)&Bs[0][r][seg * 8];
  }

  auto stage = [&](int buf, int k0) {
#pragma unroll
    for (int i = 0; i < APT; ++i)
      async_load_b128(lA[i] + buf * AbufStride, gA[i] + k0);
#pragma unroll
    for (int i = 0; i < BPT; ++i)
      async_load_b128(lB[i] + buf * BbufStride, gB[i] + k0);
  };

  FragC acc[2][2];
#pragma unroll
  for (int i = 0; i < 2; ++i)
#pragma unroll
    for (int j = 0; j < 2; ++j)
#pragma unroll
      for (int v = 0; v < 8; ++v) acc[i][j].f[v] = 0.0f;

  const int nsteps = K >> 5;
  stage(0, 0);
  for (int s = 0; s < nsteps; ++s) {
    wait_async0();          // this wave's stage(s) transfers complete
    __syncthreads();        // everyone's transfers complete; prior reads done
    if (s + 1 < nsteps) stage((s + 1) & 1, (s + 1) * 32);

    const int buf = s & 1;
    FragAB a[2], b[2];
#pragma unroll
    for (int i = 0; i < 2; ++i) {
      const unsigned short* p = &As[buf][wm * 32 + 16 * i + laneLo][8 * laneHi];
      uint4 lo = *(const uint4*)p;          // K = 8*laneHi + 0..7
      uint4 hi = *(const uint4*)(p + 16);   // K = 8*laneHi + 16..23
      a[i].u[0] = lo.x; a[i].u[1] = lo.y; a[i].u[2] = lo.z; a[i].u[3] = lo.w;
      a[i].u[4] = hi.x; a[i].u[5] = hi.y; a[i].u[6] = hi.z; a[i].u[7] = hi.w;
    }
#pragma unroll
    for (int j = 0; j < 2; ++j) {
      const unsigned short* p = &Bs[buf][wn * 32 + 16 * j + laneLo][8 * laneHi];
      uint4 lo = *(const uint4*)p;          // K = 8*laneHi + 0..7
      uint4 hi = *(const uint4*)(p + 16);   // K = 8*laneHi + 16..23
      b[j].u[0] = lo.x; b[j].u[1] = lo.y; b[j].u[2] = lo.z; b[j].u[3] = lo.w;
      b[j].u[4] = hi.x; b[j].u[5] = hi.y; b[j].u[6] = hi.z; b[j].u[7] = hi.w;
    }
#pragma unroll
    for (int i = 0; i < 2; ++i)
#pragma unroll
      for (int j = 0; j < 2; ++j)
        acc[i][j].v = __builtin_amdgcn_wmma_f32_16x16x32_bf16(
            false, a[i].v, false, b[j].v, (short)0, acc[i][j].v, false, false);
  }

#pragma unroll
  for (int i = 0; i < 2; ++i)
#pragma unroll
    for (int j = 0; j < 2; ++j)
#pragma unroll
      for (int v = 0; v < 8; ++v) {
        const int m = m0 + wm * 32 + 16 * i + v + 8 * laneHi;
        const int n = n0 + wn * 32 + 16 * j + laneLo;
        float val = acc[i][j].f[v] * alpha;
        if (bias)   val += bias[n];
        if (addsrc) val += addsrc[coff + (long long)m * ldc + n];
        if (pos)    val += pos[(m & 255) * N + n];
        if (relu)   val = val > 0.0f ? val : 0.0f;
        C[coff + (long long)m * ldc + n] = val;
      }
}

// ---------------------------------------------------------------------------
// Utility / elementwise kernels
// ---------------------------------------------------------------------------
__global__ void zero_f32_kernel(float* p, long long n) {
  long long i = (long long)blockIdx.x * blockDim.x + threadIdx.x;
  if (i < n) p[i] = 0.0f;
}

__global__ void f32_to_bf16_kernel(const float* __restrict__ in,
                                   unsigned short* __restrict__ out,
                                   long long n, int gelu) {
  long long i = (long long)blockIdx.x * blockDim.x + threadIdx.x;
  if (i >= n) return;
  float v = in[i];
  if (gelu) v = gelu_tanh(v);
  out[i] = f2bf(v);
}

// in: K x N (f32)  ->  out: N x K (bf16)
__global__ void transpose_bf16_kernel(const float* __restrict__ in,
                                      unsigned short* __restrict__ out,
                                      int K, int N) {
  long long i = (long long)blockIdx.x * blockDim.x + threadIdx.x;
  if (i >= (long long)K * N) return;
  int k = (int)(i / N), n = (int)(i % N);
  out[(long long)n * K + k] = f2bf(in[i]);
}

// qkv f32 (B,256,3,12,64) -> V^T bf16 (B,12,64,256)
__global__ void vT_bf16_kernel(const float* __restrict__ qkv,
                               unsigned short* __restrict__ vt) {
  long long i = (long long)blockIdx.x * blockDim.x + threadIdx.x;
  if (i >= 16LL * 12 * 64 * 256) return;
  int key = (int)(i & 255);
  long long q = i >> 8;
  int d = (int)(q & 63);
  long long t = q >> 6;
  int h = (int)(t % 12);
  int b = (int)(t / 12);
  float v = qkv[((long long)b * 256 + key) * 2304 + 1536 + h * 64 + d];
  vt[i] = f2bf(v);
}

// ---------------------------------------------------------------------------
// Stage kernels
// ---------------------------------------------------------------------------
__global__ void rasterize_kernel(const float* __restrict__ lm, float* mask) {
  long long i = (long long)blockIdx.x * blockDim.x + threadIdx.x;
  if (i >= 16LL * 117 * 512) return;
  int k = (int)(i % 512);
  int t = (int)(i / 512);
  int s = t % 117, b = t / 117;
  int s0 = (s < 43) ? s : ((s < 92) ? s + 1 : s + 2);
  const float* L = lm + ((long long)b * 120 + s0) * 2;
  int p1x = imin(imax((int)(L[0] * 512.0f), 0), 511);
  int p1y = imin(imax((int)(L[1] * 512.0f), 0), 511);
  int p2x = imin(imax((int)(L[2] * 512.0f), 0), 511);
  int p2y = imin(imax((int)(L[3] * 512.0f), 0), 511);
  int dy = p2y - p1y, dx = p2x - p1x;
  if (k < dy && k < dx) {
    int yi = imin(imax(p1y + k, 0), 511);
    int xi = imin(imax(p1x + k, 0), 511);
    mask[((long long)b * 512 + yi) * 512 + xi] = 1.0f;  // max(.,1) == store 1
  }
}

__global__ void pool32_kernel(const float* __restrict__ mask, float* pooled) {
  int i = blockIdx.x * blockDim.x + threadIdx.x;
  if (i >= 16 * 1024) return;
  int b = i >> 10, oy = (i >> 5) & 31, ox = i & 31;
  float s = 0.0f;
  for (int iy = 0; iy < 16; ++iy)
    for (int ix = 0; ix < 16; ++ix)
      s += mask[((long long)b * 512 + oy * 16 + iy) * 512 + ox * 16 + ix];
  pooled[i] = s * (1.0f / 256.0f);
}

__global__ void stn_kernel(const float* __restrict__ pooled,
                           const float* __restrict__ w1, const float* __restrict__ b1,
                           const float* __restrict__ w2, const float* __restrict__ b2,
                           float* theta) {
  __shared__ float h[50];
  int b = blockIdx.x;
  for (int j = threadIdx.x; j < 50; j += blockDim.x) {
    float s = b1[j];
    for (int i = 0; i < 1024; ++i) s += pooled[b * 1024 + i] * w1[i * 50 + j];
    h[j] = tanhf(s);
  }
  __syncthreads();
  for (int t = threadIdx.x; t < 6; t += blockDim.x) {
    float s = b2[t];
    for (int j = 0; j < 50; ++j) s += h[j] * w2[j * 6 + t];
    theta[b * 6 + t] = s;
  }
}

__device__ __forceinline__ float sample_one(const float* base, int yy, int xx) {
  bool ok = (yy >= 0) & (yy < 512) & (xx >= 0) & (xx < 512);
  int yc = imin(imax(yy, 0), 511), xc = imin(imax(xx, 0), 511);
  return ok ? base[(long long)yc * 512 + xc] : 0.0f;
}

__global__ void sample_kernel(const float* __restrict__ img,
                              const float* __restrict__ mask,
                              const float* __restrict__ theta,
                              float* out_img, float* out_mask) {
  long long i = (long long)blockIdx.x * blockDim.x + threadIdx.x;
  if (i >= 16LL * 512 * 512) return;
  int x = (int)(i % 512);
  int y = (int)((i / 512) % 512);
  int b = (int)(i / (512 * 512));
  const float* T = theta + b * 6;
  float gx = (2.0f * x + 1.0f) * (1.0f / 512.0f) - 1.0f;
  float gy = (2.0f * y + 1.0f) * (1.0f / 512.0f) - 1.0f;
  float u = T[0] * gx + T[1] * gy + T[2];
  float v = T[3] * gx + T[4] * gy + T[5];
  float fx = ((u + 1.0f) * 512.0f - 1.0f) * 0.5f;
  float fy = ((v + 1.0f) * 512.0f - 1.0f) * 0.5f;
  float x0f = floorf(fx), y0f = floorf(fy);
  int x0 = (int)x0f, y0 = (int)y0f;
  float wx1 = fx - x0f, wy1 = fy - y0f;
  float w00 = (1.0f - wy1) * (1.0f - wx1), w01 = (1.0f - wy1) * wx1;
  float w10 = wy1 * (1.0f - wx1),          w11 = wy1 * wx1;

  const float* mb = mask + (long long)b * 512 * 512;
  out_mask[i] = sample_one(mb, y0, x0) * w00 + sample_one(mb, y0, x0 + 1) * w01 +
                sample_one(mb, y0 + 1, x0) * w10 + sample_one(mb, y0 + 1, x0 + 1) * w11;
  for (int c = 0; c < 3; ++c) {
    const float* ib = img + ((long long)b * 3 + c) * 512 * 512;
    out_img[((long long)b * 3 + c) * 512 * 512 + (long long)y * 512 + x] =
        sample_one(ib, y0, x0) * w00 + sample_one(ib, y0, x0 + 1) * w01 +
        sample_one(ib, y0 + 1, x0) * w10 + sample_one(ib, y0 + 1, x0 + 1) * w11;
  }
}

// transformed image (B,3,512,512) -> patch matrix (4096 x 3072) in bf16
__global__ void im2col_kernel(const float* __restrict__ timg,
                              unsigned short* __restrict__ out) {
  long long i = (long long)blockIdx.x * blockDim.x + threadIdx.x;
  if (i >= 4096LL * 3072) return;
  long long m = i / 3072;
  int kk = (int)(i % 3072);
  int b = (int)(m >> 8), p = (int)(m & 255);
  int gy = p >> 4, gx = p & 15;
  int c = kk / 1024, r = kk % 1024;
  int ky = r >> 5, kx = r & 31;
  float v = timg[(((long long)b * 3 + c) * 512 + gy * 32 + ky) * 512 + gx * 32 + kx];
  out[i] = f2bf(v);
}

__global__ void layernorm_bf16_kernel(const float* __restrict__ in,
                                      const float* __restrict__ g,
                                      const float* __restrict__ bta,
                                      unsigned short* __restrict__ out, int ncol) {
  __shared__ float red[256];
  int row = blockIdx.x;
  const float* x = in + (long long)row * ncol;
  float s = 0.0f;
  for (int c = threadIdx.x; c < ncol; c += 256) s += x[c];
  red[threadIdx.x] = s; __syncthreads();
  for (int o = 128; o > 0; o >>= 1) {
    if (threadIdx.x < o) red[threadIdx.x] += red[threadIdx.x + o];
    __syncthreads();
  }
  float mean = red[0] / ncol; __syncthreads();
  float s2 = 0.0f;
  for (int c = threadIdx.x; c < ncol; c += 256) { float d = x[c] - mean; s2 += d * d; }
  red[threadIdx.x] = s2; __syncthreads();
  for (int o = 128; o > 0; o >>= 1) {
    if (threadIdx.x < o) red[threadIdx.x] += red[threadIdx.x + o];
    __syncthreads();
  }
  float inv = rsqrtf(red[0] / ncol + 1e-5f);
  for (int c = threadIdx.x; c < ncol; c += 256)
    out[(long long)row * ncol + c] = f2bf((x[c] - mean) * inv * g[c] + bta[c]);
}

__global__ void softmax256_kernel(float* x) {
  __shared__ float red[256];
  long long row = blockIdx.x;
  float* p = x + row * 256;
  int tid = threadIdx.x;
  float v = p[tid];
  red[tid] = v; __syncthreads();
  for (int o = 128; o > 0; o >>= 1) {
    if (tid < o) red[tid] = fmaxf(red[tid], red[tid + o]);
    __syncthreads();
  }
  float mx = red[0]; __syncthreads();
  float e = __expf(v - mx);
  red[tid] = e; __syncthreads();
  for (int o = 128; o > 0; o >>= 1) {
    if (tid < o) red[tid] += red[tid + o];
    __syncthreads();
  }
  p[tid] = e / red[0];
}

// tok3 (B,256,768) -> feats (B,6,768): bilinear resize box to 16x16 then mean
__global__ void roi_kernel(const float* __restrict__ tok, float* feats) {
  const int boxes[6][4] = {{0,6,0,8},{4,11,0,8},{9,16,0,8},
                           {0,6,8,16},{4,11,8,16},{9,16,8,16}};
  int i = blockIdx.x * blockDim.x + threadIdx.x;
  if (i >= 16 * 6 * 768) return;
  int ch = i % 768;
  int t = i / 768;
  int roi = t % 6, b = t / 6;
  int r0 = boxes[roi][0], r1 = boxes[roi][1], c0 = boxes[roi][2], c1 = boxes[roi][3];
  int hin = r1 - r0, win = c1 - c0;
  float sh = (float)hin / 16.0f, sw = (float)win / 16.0f;
  float acc = 0.0f;
  for (int oy = 0; oy < 16; ++oy) {
    float fy = fminf(fmaxf((oy + 0.5f) * sh - 0.5f, 0.0f), (float)hin - 1.0f);
    int y0 = (int)floorf(fy); int y1 = imin(y0 + 1, hin - 1); float wy = fy - y0;
    for (int ox = 0; ox < 16; ++ox) {
      float fx = fminf(fmaxf((ox + 0.5f) * sw - 0.5f, 0.0f), (float)win - 1.0f);
      int x0 = (int)floorf(fx); int x1 = imin(x0 + 1, win - 1); float wx = fx - x0;
      float v00 = tok[((long long)b * 256 + (r0 + y0) * 16 + (c0 + x0)) * 768 + ch];
      float v01 = tok[((long long)b * 256 + (r0 + y0) * 16 + (c0 + x1)) * 768 + ch];
      float v10 = tok[((long long)b * 256 + (r0 + y1) * 16 + (c0 + x0)) * 768 + ch];
      float v11 = tok[((long long)b * 256 + (r0 + y1) * 16 + (c0 + x1)) * 768 + ch];
      acc += v00 * (1 - wy) * (1 - wx) + v01 * (1 - wy) * wx +
             v10 * wy * (1 - wx) + v11 * wy * wx;
    }
  }
  feats[i] = acc * (1.0f / 256.0f);
}

__global__ void fc_kernel(const float* __restrict__ feats,
                          const float* __restrict__ w, const float* __restrict__ bc,
                          float* out) {
  int i = blockIdx.x * blockDim.x + threadIdx.x;
  if (i >= 96) return;   // B*6 rows
  float l[4];
  for (int c = 0; c < 4; ++c) {
    float s = bc[c];
    for (int e = 0; e < 768; ++e) s += feats[(long long)i * 768 + e] * w[e * 4 + c];
    l[c] = s;
  }
  float mx = fmaxf(fmaxf(l[0], l[1]), fmaxf(l[2], l[3]));
  float es = 0.0f;
  for (int c = 0; c < 4; ++c) { l[c] = __expf(l[c] - mx); es += l[c]; }
  for (int c = 0; c < 4; ++c) out[i * 4 + c] = l[c] / es;
}

// ---------------------------------------------------------------------------
// Host orchestration
// ---------------------------------------------------------------------------
static inline long long cdiv(long long a, long long b) { return (a + b - 1) / b; }

template <int WMW, int WNW>
static void launch_gemm_t(hipStream_t stream,
                          const unsigned short* A, long long sA1, long long sA2, int lda,
                          const unsigned short* Bt, long long sB1, long long sB2, int ldb,
                          float* C, long long sC1, long long sC2, int ldc,
                          const float* bias, const float* addsrc, const float* pos,
                          int M, int N, int K, int nz, int zdiv, float alpha, int relu) {
  dim3 grid((unsigned)((M / (WMW * 32)) * (N / (WNW * 32))), 1, (unsigned)nz);
  gemm_bf16_lds_kernel<WMW, WNW><<<grid, 256, 0, stream>>>(
      A, sA1, sA2, lda, Bt, sB1, sB2, ldb, C, sC1, sC2, ldc,
      bias, addsrc, pos, M, N, K, zdiv, alpha, relu);
}

extern "C" void kernel_launch(void* const* d_in, const int* in_sizes, int n_in,
                              void* d_out, int out_size, void* d_ws, size_t ws_size,
                              hipStream_t stream) {
  const float* x       = (const float*)d_in[0];
  const float* lm      = (const float*)d_in[1];
  const float* stn_w1  = (const float*)d_in[2];
  const float* stn_b1  = (const float*)d_in[3];
  const float* stn_w2  = (const float*)d_in[4];
  const float* stn_b2  = (const float*)d_in[5];
  const float* conv_w  = (const float*)d_in[6];
  const float* conv_b  = (const float*)d_in[7];
  const float* pe_w    = (const float*)d_in[8];
  const float* pe_b    = (const float*)d_in[9];
  const float* pos     = (const float*)d_in[10];
  const float* ln1_g   = (const float*)d_in[11];
  const float* ln1_b   = (const float*)d_in[12];
  const float* qkv_w   = (const float*)d_in[13];
  const float* qkv_b   = (const float*)d_in[14];
  const float* proj_w  = (const float*)d_in[15];
  const float* proj_b  = (const float*)d_in[16];
  const float* ln2_g   = (const float*)d_in[17];
  const float* ln2_b   = (const float*)d_in[18];
  const float* mlp_w1  = (const float*)d_in[19];
  const float* mlp_b1  = (const float*)d_in[20];
  const float* mlp_w2  = (const float*)d_in[21];
  const float* mlp_b2  = (const float*)d_in[22];
  const float* fc_w    = (const float*)d_in[23];
  const float* fc_b    = (const float*)d_in[24];

  float* out       = (float*)d_out;
  float* out_img   = out + 384;                       // (16,3,512,512)
  float* out_mask  = out_img + 16LL * 3 * 512 * 512;  // (16,1,512,512)

  // workspace bump allocator with reuse
  char* ws = (char*)d_ws;
  size_t off = 0;
  auto alloc = [&](size_t bytes) { size_t o = off; off = (off + bytes + 255) & ~(size_t)255; return o; };
  size_t off_mask   = alloc(16LL * 512 * 512 * 4);
  size_t off_pooled = alloc(16 * 1024 * 4);
  size_t off_theta  = alloc(16 * 6 * 4);
  size_t off_R1 = alloc(50331648);  // im2col bf16 / scores f32 / mlp-hidden f32
  size_t off_W1 = alloc(6291456);   // convW bf16 / projT bf16
  size_t off_Bb = alloc(16777216);  // X f32 / o f32
  size_t off_Cc = alloc(8388608);   // X bf16 / o bf16
  size_t off_W2 = alloc(1572864);   // peT bf16
  size_t off_Dd = alloc(12582912);  // tok f32 (residual stream)
  size_t off_Ee = alloc(6291456);   // ln1/ln2 bf16
  size_t off_W3 = alloc(4718592);   // qkvT / mlp1T bf16
  size_t off_Ff = alloc(37748736);  // qkv f32 / probs bf16 / hidden bf16
  size_t off_Gg = alloc(18874368);  // qkv bf16 / mlp2T bf16
  size_t off_Vt = alloc(6291456);   // V^T bf16 (B,12,64,256)
  size_t off_ft = alloc(16 * 6 * 768 * 4);

  float*          maskb  = (float*)(ws + off_mask);
  float*          pooled = (float*)(ws + off_pooled);
  float*          theta  = (float*)(ws + off_theta);
  unsigned short* R1u = (unsigned short*)(ws + off_R1);
  float*          R1f = (float*)(ws + off_R1);
  unsigned short* W1u = (unsigned short*)(ws + off_W1);
  float*          Bbf = (float*)(ws + off_Bb);
  unsigned short* Ccu = (unsigned short*)(ws + off_Cc);
  unsigned short* W2u = (unsigned short*)(ws + off_W2);
  float*          Ddf = (float*)(ws + off_Dd);
  unsigned short* Eeu = (unsigned short*)(ws + off_Ee);
  unsigned short* W3u = (unsigned short*)(ws + off_W3);
  float*          Fff = (float*)(ws + off_Ff);
  unsigned short* Ffu = (unsigned short*)(ws + off_Ff);
  unsigned short* Ggu = (unsigned short*)(ws + off_Gg);
  unsigned short* Vtu = (unsigned short*)(ws + off_Vt);
  float*          feats = (float*)(ws + off_ft);

  // 1) rasterize landmark segments into mask
  long long maskN = 16LL * 512 * 512;
  zero_f32_kernel<<<(unsigned)cdiv(maskN, 256), 256, 0, stream>>>(maskb, maskN);
  long long rN = 16LL * 117 * 512;
  rasterize_kernel<<<(unsigned)cdiv(rN, 256), 256, 0, stream>>>(lm, maskb);

  // 2) 32x32 pooling, STN MLP -> theta
  pool32_kernel<<<(unsigned)cdiv(16 * 1024, 256), 256, 0, stream>>>(maskb, pooled);
  stn_kernel<<<16, 64, 0, stream>>>(pooled, stn_w1, stn_b1, stn_w2, stn_b2, theta);

  // 3) affine grid-sample of image and mask (outputs 2 and 3)
  sample_kernel<<<(unsigned)cdiv(maskN, 256), 256, 0, stream>>>(x, maskb, theta, out_img, out_mask);

  // 4) conv as GEMM: im2col + (4096x3072)@(3072x1024) + bias, relu
  long long icN = 4096LL * 3072;
  im2col_kernel<<<(unsigned)cdiv(icN, 256), 256, 0, stream>>>(out_img, R1u);
  long long cwN = 1024LL * 3072;
  f32_to_bf16_kernel<<<(unsigned)cdiv(cwN, 256), 256, 0, stream>>>(conv_w, W1u, cwN, 0);
  launch_gemm_t<2, 4>(stream, R1u, 0, 0, 3072, W1u, 0, 0, 3072, Bbf, 0, 0, 1024,
                      conv_b, nullptr, nullptr, 4096, 1024, 3072, 1, 1, 1.0f, 1);

  // 5) patch embed + pos: tok = X @ pe_w + pe_b + pos
  f32_to_bf16_kernel<<<(unsigned)cdiv(4096LL * 1024, 256), 256, 0, stream>>>(Bbf, Ccu, 4096LL * 1024, 0);
  transpose_bf16_kernel<<<(unsigned)cdiv(1024LL * 768, 256), 256, 0, stream>>>(pe_w, W2u, 1024, 768);
  launch_gemm_t<2, 4>(stream, Ccu, 0, 0, 1024, W2u, 0, 0, 1024, Ddf, 0, 0, 768,
                      pe_b, nullptr, pos, 4096, 768, 1024, 1, 1, 1.0f, 0);

  // 6) LN1 -> bf16, QKV GEMM
  layernorm_bf16_kernel<<<4096, 256, 0, stream>>>(Ddf, ln1_g, ln1_b, Eeu, 768);
  transpose_bf16_kernel<<<(unsigned)cdiv(768LL * 2304, 256), 256, 0, stream>>>(qkv_w, W3u, 768, 2304);
  launch_gemm_t<2, 4>(stream, Eeu, 0, 0, 768, W3u, 0, 0, 768, Fff, 0, 0, 2304,
                      qkv_b, nullptr, nullptr, 4096, 2304, 768, 1, 1, 1.0f, 0);
  f32_to_bf16_kernel<<<(unsigned)cdiv(4096LL * 2304, 256), 256, 0, stream>>>(Fff, Ggu, 4096LL * 2304, 0);
  vT_bf16_kernel<<<(unsigned)cdiv(16LL * 12 * 64 * 256, 256), 256, 0, stream>>>(Fff, Vtu);

  // 7) attention scores: per (b,h): Q(256x64) @ K(256x64)^T, alpha = 1/sqrt(64)
  launch_gemm_t<2, 4>(stream,
                      Ggu, 589824, 64, 2304,          /* Q rows in qkv bf16 */
                      Ggu + 768, 589824, 64, 2304,    /* K rows = BT (N x K) */
                      R1f, 786432, 65536, 256,
                      nullptr, nullptr, nullptr, 256, 256, 64, 192, 12, 0.125f, 0);
  softmax256_kernel<<<192 * 256, 256, 0, stream>>>(R1f);
  f32_to_bf16_kernel<<<(unsigned)cdiv(192LL * 65536, 256), 256, 0, stream>>>(R1f, Ffu, 192LL * 65536, 0);

  // 8) AV: probs(256x256) @ V(256x64) with V pre-transposed (BT path), N=64
  launch_gemm_t<4, 2>(stream,
                      Ffu, 786432, 65536, 256,
                      Vtu, 196608, 16384, 256,
                      Bbf, 196608, 64, 768,
                      nullptr, nullptr, nullptr, 256, 64, 256, 192, 12, 1.0f, 0);

  // 9) proj + residual (in-place on tok)
  f32_to_bf16_kernel<<<(unsigned)cdiv(4096LL * 768, 256), 256, 0, stream>>>(Bbf, Ccu, 4096LL * 768, 0);
  transpose_bf16_kernel<<<(unsigned)cdiv(768LL * 768, 256), 256, 0, stream>>>(proj_w, W1u, 768, 768);
  launch_gemm_t<2, 4>(stream, Ccu, 0, 0, 768, W1u, 0, 0, 768, Ddf, 0, 0, 768,
                      proj_b, Ddf, nullptr, 4096, 768, 768, 1, 1, 1.0f, 0);

  // 10) LN2, MLP with GELU, residual
  layernorm_bf16_kernel<<<4096, 256, 0, stream>>>(Ddf, ln2_g, ln2_b, Eeu, 768);
  transpose_bf16_kernel<<<(unsigned)cdiv(768LL * 3072, 256), 256, 0, stream>>>(mlp_w1, W3u, 768, 3072);
  launch_gemm_t<2, 4>(stream, Eeu, 0, 0, 768, W3u, 0, 0, 768, R1f, 0, 0, 3072,
                      mlp_b1, nullptr, nullptr, 4096, 3072, 768, 1, 1, 1.0f, 0);
  f32_to_bf16_kernel<<<(unsigned)cdiv(4096LL * 3072, 256), 256, 0, stream>>>(R1f, Ffu, 4096LL * 3072, 1);
  transpose_bf16_kernel<<<(unsigned)cdiv(3072LL * 768, 256), 256, 0, stream>>>(mlp_w2, Ggu, 3072, 768);
  launch_gemm_t<2, 4>(stream, Ffu, 0, 0, 3072, Ggu, 0, 0, 3072, Ddf, 0, 0, 768,
                      mlp_b2, Ddf, nullptr, 4096, 768, 3072, 1, 1, 1.0f, 0);

  // 11) ROI pooling + classifier softmax (output 1)
  roi_kernel<<<(unsigned)cdiv(16 * 6 * 768, 256), 256, 0, stream>>>(Ddf, feats);
  fc_kernel<<<2, 64, 0, stream>>>(feats, fc_w, fc_b, out);

  (void)in_sizes; (void)n_in; (void)out_size; (void)ws_size;
}
